// TargetCMLILoss_51359218925723
// MI455X (gfx1250) — compile-verified, matrix operand order checked by reading, not run on hardware
//
#include <hip/hip_runtime.h>
#include <hip/hip_bf16.h>

// ---------------------------------------------------------------------------
// Problem constants (from reference): B=32, S=512, D=768; sliced len M=511,
// padded to MP=512 for 16x16 WMMA tiling. K = D = 768 = 24 steps of 32.
// ---------------------------------------------------------------------------
constexpr int BB = 32;
constexpr int SS = 512;
constexpr int DD = 768;
constexpr int MM = 511;
constexpr int MP = 512;
constexpr int KSTEPS = DD / 32;      // 24
constexpr int NTILES = MP / 16;      // 32
constexpr int LDSB_STRIDE = DD + 8;  // 776 halves -> avoids 64-bank conflicts

typedef __attribute__((ext_vector_type(16))) _Float16 v16h;
typedef __attribute__((ext_vector_type(8)))  _Float16 v8h;
typedef __attribute__((ext_vector_type(4)))  _Float16 v4h;
typedef __attribute__((ext_vector_type(8)))  float    v8f;

// ---------------------------------------------------------------------------
// Kernel 0: zero the scalar accumulators (graph-capture-safe init).
// ---------------------------------------------------------------------------
__global__ void k_zero_accum(float* accum) {
    if (threadIdx.x < 4) accum[threadIdx.x] = 0.0f;
}

// ---------------------------------------------------------------------------
// Kernel 1: L2-normalize rows of text[:,1:] and target[:,1:] -> f16 matrices
// in workspace, padded to 512 rows (row 511 = zeros). One wave per row.
// ---------------------------------------------------------------------------
__global__ void __launch_bounds__(256)
k_normalize(const float* __restrict__ text, const float* __restrict__ target,
            _Float16* __restrict__ txtn, _Float16* __restrict__ tgtn) {
    int wid  = (blockIdx.x * blockDim.x + threadIdx.x) >> 5;
    int lane = threadIdx.x & 31;
    int mat  = wid / (BB * MP);
    int rem  = wid % (BB * MP);
    int b    = rem / MP;
    int row  = rem % MP;

    _Float16* dst = (mat ? tgtn : txtn) + ((size_t)b * MP + row) * DD;

    if (row >= MM) {  // zero padding row
        v4h z = {};
        #pragma unroll
        for (int j = 0; j < 6; ++j) *(v4h*)(dst + (lane + 32 * j) * 4) = z;
        return;
    }

    const float* src = (mat ? target : text) + ((size_t)b * SS + row + 1) * DD;
    float4 v[6];
    float ss = 0.0f;
    #pragma unroll
    for (int j = 0; j < 6; ++j) {
        v[j] = *(const float4*)(src + (lane + 32 * j) * 4);
        ss += v[j].x * v[j].x + v[j].y * v[j].y + v[j].z * v[j].z + v[j].w * v[j].w;
    }
    #pragma unroll
    for (int m = 16; m; m >>= 1) ss += __shfl_xor(ss, m, 32);
    float rinv = rsqrtf(ss);
    #pragma unroll
    for (int j = 0; j < 6; ++j) {
        v4h h;
        h.x = (_Float16)(v[j].x * rinv);
        h.y = (_Float16)(v[j].y * rinv);
        h.z = (_Float16)(v[j].z * rinv);
        h.w = (_Float16)(v[j].w * rinv);
        *(v4h*)(dst + (lane + 32 * j) * 4) = h;
    }
}

// ---------------------------------------------------------------------------
// Kernel 2: fused batched GEMM (sim = txt_n . tgt_n^T) + per-row argmax.
// Block = 128 threads = 4 waves; each wave owns one 16-row M-tile; the 4
// waves share each 16-col B-tile, DOUBLE-BUFFERED in LDS via CDNA5 async
// global->LDS copies (global_load_async_to_lds_b128, ASYNCcnt-tracked), so
// the next tile's staging overlaps the current tile's 24 WMMAs. sim is never
// materialized. Argmax indices -> idxbuf (int) and d_out[3..] (float).
// ---------------------------------------------------------------------------
__global__ void __launch_bounds__(128)
k_sim_argmax(const _Float16* __restrict__ txtn, const _Float16* __restrict__ tgtn,
             int* __restrict__ idxbuf, float* __restrict__ out_idx) {
    __shared__ _Float16 Bs[2][16 * LDSB_STRIDE];   // 2 x 24.25 KB

    const int b      = blockIdx.x >> 3;
    const int wave   = threadIdx.x >> 5;
    const int lane   = threadIdx.x & 31;
    const int m_tile = ((blockIdx.x & 7) << 2) + wave;

    // A fragment base: lane L<16 -> row L of tile, K chunks {k..k+7, k+16..k+23};
    // lanes 16-31 shifted by +8 in K (per CDNA5 16-bit A layout).
    const _Float16* Abase =
        txtn + ((size_t)b * MP + m_tile * 16 + (lane & 15)) * DD + ((lane >> 4) << 3);
    const _Float16* Bbat = tgtn + (size_t)b * MP * DD;

    // Async-stage one 16x768 B tile (1536 x 16B chunks, 12 per thread) into
    // LDS buffer `buf`. Flat shared-pointer low 32 bits == LDS byte offset.
    auto stage_async = [&](int nt, int buf) {
        #pragma unroll
        for (int i = 0; i < 12; ++i) {
            int c   = threadIdx.x + 128 * i;
            int row = c / 96;
            int off = (c % 96) * 8;
            const _Float16* g = Bbat + (size_t)(nt * 16 + row) * DD + off;
            unsigned lds_off  = (unsigned)(size_t)&Bs[buf][row * LDSB_STRIDE + off];
            asm volatile("global_load_async_to_lds_b128 %0, %1, off"
                         :: "v"(lds_off), "v"(g) : "memory");
        }
    };

    float best[8];
    int   bestn[8];
    #pragma unroll
    for (int r = 0; r < 8; ++r) { best[r] = -3.4e38f; bestn[r] = 0; }

    stage_async(0, 0);
    asm volatile("s_wait_asynccnt 0x0" ::: "memory");
    __syncthreads();

    for (int nt = 0; nt < NTILES; ++nt) {
        const int cur = nt & 1;
        if (nt + 1 < NTILES) stage_async(nt + 1, cur ^ 1);  // overlap with WMMAs

        v8f acc = {};
        const _Float16* Bl = &Bs[cur][(lane & 15) * LDSB_STRIDE + ((lane >> 4) << 3)];
        #pragma unroll 8
        for (int ks = 0; ks < KSTEPS; ++ks) {
            v16h a, bf;
            const _Float16* ap = Abase + ks * 32;
            *(v8h*)&a        = *(const v8h*)ap;          // global_load_b128
            *((v8h*)&a + 1)  = *(const v8h*)(ap + 16);
            const _Float16* bp = Bl + ks * 32;
            *(v8h*)&bf       = *(const v8h*)bp;          // ds_load_b128
            *((v8h*)&bf + 1) = *(const v8h*)(bp + 16);
            acc = __builtin_amdgcn_wmma_f32_16x16x32_f16(
                false, a, false, bf, (short)0, acc, false, false);
        }

        int n = nt * 16 + (lane & 15);
        if (n < MM) {   // mask out padded column 511 (its sim==0 could win)
            #pragma unroll
            for (int r = 0; r < 8; ++r) {
                float v = acc[r];
                if (v > best[r]) { best[r] = v; bestn[r] = n; }
            }
        }

        // Own async loads done + all waves done reading `cur` before reuse.
        asm volatile("s_wait_asynccnt 0x0" ::: "memory");
        __syncthreads();
    }

    // Cross-lane argmax within each 16-lane half (rows r / r+8), preferring
    // the lowest column index on ties (matches jnp.argmax).
    #pragma unroll
    for (int r = 0; r < 8; ++r) {
        float v = best[r];
        int   n = bestn[r];
        #pragma unroll
        for (int m = 8; m; m >>= 1) {
            float v2 = __shfl_xor(v, m, 32);
            int   n2 = __shfl_xor(n, m, 32);
            if (v2 > v || (v2 == v && n2 < n)) { v = v2; n = n2; }
        }
        best[r] = v; bestn[r] = n;
    }

    if ((lane & 15) == 0) {
        int rbase = m_tile * 16 + ((lane >> 4) << 3);
        #pragma unroll
        for (int r = 0; r < 8; ++r) {
            int row = rbase + r;
            if (row < MM) {
                idxbuf[b * MM + row]  = bestn[r];
                out_idx[b * MM + row] = (float)bestn[r];
            }
        }
    }
}

// ---------------------------------------------------------------------------
// Kernel 3: masked text loss. One wave per (b, m) row: sum over D of
// (txt - target[b, idx+1])^2, gated by padding_mask==0; atomic accumulate.
// ---------------------------------------------------------------------------
__global__ void __launch_bounds__(256)
k_textloss(const float* __restrict__ text, const float* __restrict__ target,
           const int* __restrict__ pad, const int* __restrict__ idxbuf,
           float* __restrict__ accum) {
    int wid  = (blockIdx.x * blockDim.x + threadIdx.x) >> 5;
    int lane = threadIdx.x & 31;
    if (wid >= BB * MM) return;
    int b = wid / MM, m = wid % MM;
    int valid = (pad[b * SS + m + 1] == 0);
    int idx   = idxbuf[wid];

    const float* trow = text   + ((size_t)b * SS + m + 1) * DD;
    const float* arow = target + ((size_t)b * SS + idx + 1) * DD;
    float s = 0.0f;
    #pragma unroll
    for (int j = 0; j < 6; ++j) {
        float4 t = *(const float4*)(trow + (lane + 32 * j) * 4);
        float4 a = *(const float4*)(arow + (lane + 32 * j) * 4);
        float dx = t.x - a.x, dy = t.y - a.y, dz = t.z - a.z, dw = t.w - a.w;
        s += dx * dx + dy * dy + dz * dz + dw * dw;
    }
    #pragma unroll
    for (int mm = 16; mm; mm >>= 1) s += __shfl_xor(s, mm, 32);
    if (lane == 0) {
        if (valid) atomicAdd(&accum[0], s);
        atomicAdd(&accum[1], (float)valid);
    }
}

// ---------------------------------------------------------------------------
// Kernel 4: image MSE partial sums (grid-stride float4 stream).
// ---------------------------------------------------------------------------
__global__ void __launch_bounds__(256)
k_imgloss(const float* __restrict__ image, const float* __restrict__ target,
          float* __restrict__ accum, int n4) {
    int tid    = blockIdx.x * blockDim.x + threadIdx.x;
    int stride = gridDim.x * blockDim.x;
    float s = 0.0f;
    for (int i = tid; i < n4; i += stride) {
        float4 a = ((const float4*)image)[i];
        float4 t = ((const float4*)target)[i];
        float dx = a.x - t.x, dy = a.y - t.y, dz = a.z - t.z, dw = a.w - t.w;
        s += dx * dx + dy * dy + dz * dz + dw * dw;
    }
    #pragma unroll
    for (int m = 16; m; m >>= 1) s += __shfl_xor(s, m, 32);
    if ((threadIdx.x & 31) == 0) atomicAdd(&accum[2], s);
}

// ---------------------------------------------------------------------------
// Kernel 5: finalize scalars into d_out[0..2].
// ---------------------------------------------------------------------------
__global__ void k_finalize(const float* __restrict__ accum, float* __restrict__ out) {
    if (threadIdx.x == 0) {
        float text_loss = accum[0] / (accum[1] * (float)DD);
        float img_loss  = accum[2] / (float)((size_t)BB * SS * DD);
        out[0] = 0.5f * (text_loss + img_loss);
        out[1] = text_loss;
        out[2] = img_loss;
    }
}

// ---------------------------------------------------------------------------
extern "C" void kernel_launch(void* const* d_in, const int* in_sizes, int n_in,
                              void* d_out, int out_size, void* d_ws, size_t ws_size,
                              hipStream_t stream) {
    const float* image   = (const float*)d_in[0];
    const float* text    = (const float*)d_in[1];
    const float* target  = (const float*)d_in[2];
    const int*   padmask = (const int*)d_in[3];
    float*       out     = (float*)d_out;

    // Workspace layout
    const size_t HALF_ELEMS = (size_t)BB * MP * DD;           // 12,582,912
    _Float16* txtn  = (_Float16*)d_ws;
    _Float16* tgtn  = txtn + HALF_ELEMS;
    float*    accum = (float*)((char*)d_ws + 2 * HALF_ELEMS * sizeof(_Float16));
    int*      idxb  = (int*)((char*)accum + 64);

    k_zero_accum<<<1, 32, 0, stream>>>(accum);

    {   // 2 matrices * B * MP rows, one wave each; 8 waves per block
        int waves = 2 * BB * MP;
        k_normalize<<<waves / 8, 256, 0, stream>>>(text, target, txtn, tgtn);
    }

    // 8 blocks per batch (4 M-tiles each) -> 256 blocks of 128 threads
    k_sim_argmax<<<BB * 8, 128, 0, stream>>>(txtn, tgtn, idxb, out + 3);

    {   // one wave per (b, m) row
        int waves = BB * MM;                       // 16,352
        k_textloss<<<(waves + 7) / 8, 256, 0, stream>>>(text, target, padmask, idxb, accum);
    }

    {   // image MSE over B*S*D floats as float4s
        int n4 = (BB * SS * DD) / 4;               // 3,145,728
        k_imgloss<<<2048, 256, 0, stream>>>(image, target, accum, n4);
    }

    k_finalize<<<1, 32, 0, stream>>>(accum, out);
}
